// Decoder_21036749816351
// MI455X (gfx1250) — matrix operand order repaired and checked by reading, not compiled, not run on hardware
//
#include <hip/hip_runtime.h>
#include <math.h>

typedef __attribute__((ext_vector_type(16))) _Float16 v16h;
typedef __attribute__((ext_vector_type(8)))  _Float16 v8h;
typedef __attribute__((ext_vector_type(8)))  float    v8f;

#define NPRI   8732
#define NLOC   1940
#define BATCH  16
#define MAXOBJ 10
#define ZWHAT  64
#define IMG    300
#define DEC    64

// prior index -> location index (from _recon_indices)
__device__ __forceinline__ int prior_loc(int p) {
    if (p < 5776) return p >> 2;                      // 38x38, 4/loc
    if (p < 7942) return 1444 + (p - 5776) / 6;       // 19x19, 6/loc
    if (p < 8542) return 1805 + (p - 7942) / 6;       // 10x10, 6/loc
    if (p < 8692) return 1905 + (p - 8542) / 6;       // 5x5,   6/loc
    if (p < 8728) return 1930 + ((p - 8692) >> 2);    // 3x3,   4/loc
    return 1939;                                      // 1x1
}

// ---------------------------------------------------------------------------
// Stage 1+2: per-batch stable top-10 by depth, gather z_what (as f16) etc.
// ---------------------------------------------------------------------------
__global__ __launch_bounds__(256)
void topk_gather_kernel(const float* __restrict__ z_what,
                        const float* __restrict__ z_where,
                        const int*   __restrict__ z_present,
                        const float* __restrict__ z_depth,
                        _Float16* __restrict__ sw,   // (160,64) f16
                        float* __restrict__ swr,     // (160,4)
                        int*   __restrict__ sp) {    // (160,)
    __shared__ float zds[NPRI];
    __shared__ float rv[256];
    __shared__ int   ri[256];
    __shared__ int   chosen[MAXOBJ];
    __shared__ int   cloc[MAXOBJ];
    const int b = blockIdx.x;
    const int tid = threadIdx.x;

    for (int p = tid; p < NPRI; p += 256)
        zds[p] = z_depth[b * NLOC + prior_loc(p)];
    __syncthreads();

    for (int round = 0; round < MAXOBJ; ++round) {
        float bv = -INFINITY; int bi = -1;
        for (int p = tid; p < NPRI; p += 256) {
            float v = zds[p];
            if (v > bv) { bv = v; bi = p; }  // strict > keeps lowest index on ties
        }
        rv[tid] = bv; ri[tid] = bi;
        __syncthreads();
        for (int s = 128; s > 0; s >>= 1) {
            if (tid < s) {
                float v2 = rv[tid + s]; int i2 = ri[tid + s];
                bool take = (v2 > rv[tid]) ||
                            (v2 == rv[tid] && i2 >= 0 && (ri[tid] < 0 || i2 < ri[tid]));
                if (take) { rv[tid] = v2; ri[tid] = i2; }
            }
            __syncthreads();
        }
        if (tid == 0) { chosen[round] = ri[0]; zds[ri[0]] = -INFINITY; }
        __syncthreads();
    }
    if (tid < MAXOBJ) cloc[tid] = prior_loc(chosen[tid]);
    __syncthreads();

    for (int e = tid; e < MAXOBJ * ZWHAT; e += 256) {
        int k = e >> 6, c = e & 63;
        sw[(b * MAXOBJ + k) * ZWHAT + c] =
            (_Float16)z_what[(b * NLOC + cloc[k]) * ZWHAT + c];
    }
    if (tid < MAXOBJ * 4) {
        int k = tid >> 2, j = tid & 3;
        swr[(b * MAXOBJ + k) * 4 + j] = z_where[(b * NPRI + chosen[k]) * 4 + j];
    }
    if (tid < MAXOBJ)
        sp[b * MAXOBJ + tid] = z_present[b * NPRI + chosen[tid]];
}

// ---------------------------------------------------------------------------
// Weight pack: W (C, O, 2, 2) f32 row-major == (K, NC) -> transposed, padded
// f16 buffer Wb[(j, k)] with j < NCpad (zero past NC), k < Kpad (zero past C).
// ---------------------------------------------------------------------------
__global__ __launch_bounds__(256)
void pack_weights_kernel(const float* __restrict__ Wsrc, _Float16* __restrict__ Wdst,
                         int C, int NC, int NCpad, int Kpad) {
    int i = blockIdx.x * 256 + threadIdx.x;
    if (i >= NCpad * Kpad) return;
    int j = i / Kpad, k = i - j * Kpad;
    Wdst[i] = (j < NC && k < C) ? (_Float16)Wsrc[k * NC + j] : (_Float16)0.f;
}

// ---------------------------------------------------------------------------
// Stage 3: 2x2 stride-2 deconv as WMMA GEMM, all shapes compile-time.
//   X : f16 NHWC, row m=(n,y,x) stride C          (layer 0: (160,64))
//   Wb: f16 (NCpad, Kpad) transposed packed weights
//   Y : f16 NHWC (relu) or — FINAL — f32 NCHW (sigmoid) for the STN
// One wave computes 16 rows x (16*NT) cols; A fragment reused across NT WMMAs.
// ---------------------------------------------------------------------------
template <int C, int O, int H, int W, int NT, bool FINAL>
__global__ __launch_bounds__(32)
void deconv_gemm_kernel(const _Float16* __restrict__ X,
                        const _Float16* __restrict__ Wb,
                        const float* __restrict__ bias,
                        void* __restrict__ Yv) {
    constexpr int HW     = H * W;
    constexpr int NC     = 4 * O;
    constexpr int Kpad   = ((C + 31) / 32) * 32;
    constexpr int ksteps = Kpad / 32;
    const int lane = threadIdx.x;           // wave32
    const int half = lane >> 4;             // 0: lanes 0-15, 1: lanes 16-31
    const int l16  = lane & 15;
    const int tileM = blockIdx.x;
    const int tileN = blockIdx.y;

    // A row for this lane (M = lane%16 within tile)
    const int m = tileM * 16 + l16;
    const _Float16* Arow = X + m * C;

    // B columns for the NT tiles
    const _Float16* Bcol[NT];
#pragma unroll
    for (int t = 0; t < NT; ++t)
        Bcol[t] = Wb + (tileN * (16 * NT) + t * 16 + l16) * Kpad;

    v8f acc[NT];
#pragma unroll
    for (int t = 0; t < NT; ++t) acc[t] = (v8f){0.f,0.f,0.f,0.f,0.f,0.f,0.f,0.f};

#pragma unroll
    for (int ks = 0; ks < ksteps; ++ks) {
        const int kb = ks * 32;
        // A 16x32 f16 fragment: half h owns K = kb + {0..7, 16..23} + 8*h
        v16h a;
        if (kb + 16 <= C) {                       // constant after unroll
            v8h lo = *(const v8h*)(Arow + kb + half * 8);
#pragma unroll
            for (int e = 0; e < 8; ++e) a[e] = lo[e];
        } else {
#pragma unroll
            for (int e = 0; e < 8; ++e) a[e] = (_Float16)0.f;
        }
        if (kb + 32 <= C) {
            v8h hi = *(const v8h*)(Arow + kb + 16 + half * 8);
#pragma unroll
            for (int e = 0; e < 8; ++e) a[8 + e] = hi[e];
        } else {
#pragma unroll
            for (int e = 0; e < 8; ++e) a[8 + e] = (_Float16)0.f;
        }
#pragma unroll
        for (int t = 0; t < NT; ++t) {
            // B 32x16 f16 fragment: half h owns K = kb + {0..7, 16..23} + 8*h (padded buf)
            v8h b0 = *(const v8h*)(Bcol[t] + kb + half * 8);
            v8h b1 = *(const v8h*)(Bcol[t] + kb + 16 + half * 8);
            v16h bf;
#pragma unroll
            for (int e = 0; e < 8; ++e) { bf[e] = b0[e]; bf[8 + e] = b1[e]; }
            acc[t] = __builtin_amdgcn_wmma_f32_16x16x32_f16(
                false, a, false, bf, (short)0, acc[t], false, false);
        }
    }

    // Store: C/D layout -> VGPR r holds row (r + 8*half), col = lane%16.
#pragma unroll
    for (int t = 0; t < NT; ++t) {
        const int j = tileN * (16 * NT) + t * 16 + l16;
        if (NC % (16 * NT) != 0 && j >= NC) continue;   // only last layer (NC=12)
        const int o  = j >> 2;
        const int aa = (j >> 1) & 1;
        const int bb = j & 1;
        const float bv = bias[o];
#pragma unroll
        for (int r = 0; r < 8; ++r) {
            const int m2 = tileM * 16 + r + half * 8;
            const int n2 = m2 / HW;
            const int rm = m2 - n2 * HW;
            const int h2 = rm / W;
            const int w2 = rm - h2 * W;
            const int y2 = 2 * h2 + aa;
            const int x2 = 2 * w2 + bb;
            float v = acc[t][r] + bv;
            if (FINAL) {
                v = 1.f / (1.f + __expf(-v));            // sigmoid
                ((float*)Yv)[((n2 * O + o) * (2 * H) + y2) * (2 * W) + x2] = v;
            } else {
                v = fmaxf(v, 0.f);                       // relu
                ((_Float16*)Yv)[((n2 * (2 * H) + y2) * (2 * W) + x2) * O + o] =
                    (_Float16)v;
            }
        }
    }
}

// ---------------------------------------------------------------------------
// Stage 4: fused STN bilinear sampling + z-ordered per-channel compositing.
// decoded: (160, 3, 64, 64) f32; out: (16, 3, 300, 300) f32
// ---------------------------------------------------------------------------
__global__ __launch_bounds__(256)
void stn_composite_kernel(const float* __restrict__ decoded,
                          const float* __restrict__ swr,
                          const int*   __restrict__ sp,
                          float* __restrict__ out) {
    __shared__ float s_where[MAXOBJ][4];
    __shared__ int   s_pres[MAXOBJ];
    const int b   = blockIdx.y;
    const int tid = threadIdx.x;
    if (tid < MAXOBJ * 4) s_where[tid >> 2][tid & 3] = swr[b * MAXOBJ * 4 + tid];
    if (tid < MAXOBJ)     s_pres[tid] = sp[b * MAXOBJ + tid];
    __syncthreads();

    const int pix = blockIdx.x * blockDim.x + tid;
    if (pix >= IMG * IMG) return;
    const int py = pix / IMG;
    const int px = pix - py * IMG;
    const float xo = (2.f * px + 1.f) / IMG - 1.f;
    const float yo = (2.f * py + 1.f) / IMG - 1.f;

    float val0 = 0.f, val1 = 0.f, val2 = 0.f;
    bool  d0 = false, d1 = false, d2 = false;

    for (int k = 0; k < MAXOBJ; ++k) {
        if (!s_pres[k]) continue;
        const float cx = s_where[k][0], cy = s_where[k][1];
        const float w  = s_where[k][2], h  = s_where[k][3];
        const float u  = (xo - (2.f * cx - 1.f)) / (w + 1e-6f);
        const float v  = (yo - (2.f * cy - 1.f)) / (h + 1e-6f);
        const float ix = ((u + 1.f) * DEC - 1.f) * 0.5f;
        const float iy = ((v + 1.f) * DEC - 1.f) * 0.5f;
        const float fx0 = floorf(ix), fy0 = floorf(iy);
        const float fx1 = fx0 + 1.f,  fy1 = fy0 + 1.f;
        const float wx1 = ix - fx0,   wy1 = iy - fy0;
        const float wx0 = 1.f - wx1,  wy0 = 1.f - wy1;
        const bool vx0 = (fx0 >= 0.f) && (fx0 < (float)DEC);
        const bool vx1 = (fx1 >= 0.f) && (fx1 < (float)DEC);
        const bool vy0 = (fy0 >= 0.f) && (fy0 < (float)DEC);
        const bool vy1 = (fy1 >= 0.f) && (fy1 < (float)DEC);
        if (!((vx0 || vx1) && (vy0 || vy1))) continue;
        const int xi0 = (int)fminf(fmaxf(fx0, 0.f), (float)(DEC - 1));
        const int xi1 = (int)fminf(fmaxf(fx1, 0.f), (float)(DEC - 1));
        const int yi0 = (int)fminf(fmaxf(fy0, 0.f), (float)(DEC - 1));
        const int yi1 = (int)fminf(fmaxf(fy1, 0.f), (float)(DEC - 1));
        const float wA = (vy0 && vx0) ? wy0 * wx0 : 0.f;
        const float wB = (vy0 && vx1) ? wy0 * wx1 : 0.f;
        const float wC = (vy1 && vx0) ? wy1 * wx0 : 0.f;
        const float wD = (vy1 && vx1) ? wy1 * wx1 : 0.f;
        const float* img = decoded + (b * MAXOBJ + k) * 3 * DEC * DEC;
#pragma unroll
        for (int c = 0; c < 3; ++c) {
            const float* ch = img + c * DEC * DEC;
            float s = ch[yi0 * DEC + xi0] * wA + ch[yi0 * DEC + xi1] * wB +
                      ch[yi1 * DEC + xi0] * wC + ch[yi1 * DEC + xi1] * wD;
            if (s != 0.f) {
                if (c == 0 && !d0) { val0 = s; d0 = true; }
                if (c == 1 && !d1) { val1 = s; d1 = true; }
                if (c == 2 && !d2) { val2 = s; d2 = true; }
            }
        }
        if (d0 && d1 && d2) break;
    }
    const int obase = ((b * 3) * IMG + py) * IMG + px;
    out[obase]                 = val0;
    out[obase + IMG * IMG]     = val1;
    out[obase + 2 * IMG * IMG] = val2;
}

// ---------------------------------------------------------------------------
extern "C" void kernel_launch(void* const* d_in, const int* in_sizes, int n_in,
                              void* d_out, int out_size, void* d_ws, size_t ws_size,
                              hipStream_t stream) {
    const float* z_what    = (const float*)d_in[0];
    const float* z_where   = (const float*)d_in[1];
    const int*   z_present = (const int*)d_in[2];
    const float* z_depth   = (const float*)d_in[3];
    const float* Wl[6]; const float* bl[6];
    for (int i = 0; i < 6; ++i) {
        Wl[i] = (const float*)d_in[4 + 2 * i];
        bl[i] = (const float*)d_in[5 + 2 * i];
    }
    const size_t MB = 1024u * 1024u;
    char* ws = (char*)d_ws;
    char* bufA = ws;               // 8 MB: f16 acts / final f32 decoded (7.86 MB)
    char* bufB = ws + 8 * MB;      // 6 MB: f16 acts (max 5.24 MB, L4 out)
    _Float16* wb = (_Float16*)(ws + 14 * MB);   // packed weights (~0.47 MB)
    float* swr = (float*)(ws + 15 * MB);
    int*   sp  = (int*)  (ws + 15 * MB + 4096);
    float* outp = (float*)d_out;

    // Packed weight geometry per layer: {C, NC, NCpad (tile-width mult), Kpad}
    const int wgC[6]  = { 64, 256, 128, 64, 32, 16 };
    const int wgNC[6] = { 1024, 512, 256, 128, 64, 12 };
    const int wgNP[6] = { 1024, 512, 256, 128, 64, 16 };
    const int wgKP[6] = { 64, 256, 128, 64, 32, 32 };
    _Float16* wbp[6];
    {
        size_t off = 0;
        for (int i = 0; i < 6; ++i) {
            wbp[i] = wb + off;
            off += (size_t)wgNP[i] * wgKP[i];
        }
    }
    for (int i = 0; i < 6; ++i) {
        int n = wgNP[i] * wgKP[i];
        pack_weights_kernel<<<(n + 255) / 256, 256, 0, stream>>>(
            Wl[i], wbp[i], wgC[i], wgNC[i], wgNP[i], wgKP[i]);
    }

    // Stage 1+2: top-10 per batch + gather (f16 sw -> bufA)
    topk_gather_kernel<<<BATCH, 256, 0, stream>>>(z_what, z_where, z_present,
                                                  z_depth, (_Float16*)bufA, swr, sp);

    // Stage 3: six deconv WMMA GEMMs (ping-pong), chans 64->256->...->3
    deconv_gemm_kernel<64, 256, 1, 1, 4, false>
        <<<dim3(160 / 16, 1024 / 64), 32, 0, stream>>>(
            (const _Float16*)bufA, wbp[0], bl[0], bufB);
    deconv_gemm_kernel<256, 128, 2, 2, 4, false>
        <<<dim3(640 / 16, 512 / 64), 32, 0, stream>>>(
            (const _Float16*)bufB, wbp[1], bl[1], bufA);
    deconv_gemm_kernel<128, 64, 4, 4, 4, false>
        <<<dim3(2560 / 16, 256 / 64), 32, 0, stream>>>(
            (const _Float16*)bufA, wbp[2], bl[2], bufB);
    deconv_gemm_kernel<64, 32, 8, 8, 4, false>
        <<<dim3(10240 / 16, 128 / 64), 32, 0, stream>>>(
            (const _Float16*)bufB, wbp[3], bl[3], bufA);
    deconv_gemm_kernel<32, 16, 16, 16, 4, false>
        <<<dim3(40960 / 16, 64 / 64), 32, 0, stream>>>(
            (const _Float16*)bufA, wbp[4], bl[4], bufB);
    deconv_gemm_kernel<16, 3, 32, 32, 1, true>
        <<<dim3(163840 / 16, 1), 32, 0, stream>>>(
            (const _Float16*)bufB, wbp[5], bl[5], bufA);  // f32 NCHW decoded

    // Stage 4: fused STN + composite -> d_out (writes every output pixel)
    dim3 sgrid((IMG * IMG + 255) / 256, BATCH);
    stn_composite_kernel<<<sgrid, 256, 0, stream>>>((const float*)bufA, swr, sp, outp);
}
